// Llama4VisionAttention_9483287789690
// MI455X (gfx1250) — matrix operand order repaired and checked by reading, not compiled, not run on hardware
//
#include <hip/hip_runtime.h>

// ---------------------------------------------------------------------------
// Llama4 Vision attention for gfx1250 (MI455X): bf16 WMMA everywhere.
// B=16, S=577, D=1280, H=16, HD=80.
// ---------------------------------------------------------------------------

typedef __attribute__((ext_vector_type(16))) __bf16 v16bf;
typedef __attribute__((ext_vector_type(8)))  __bf16 v8bf;
typedef __attribute__((ext_vector_type(8)))  float  v8f;

#define B_    16
#define S_    577
#define D_    1280
#define H_    16
#define HD_   80
#define HALF_ 40
#define SP_   608   // keys/queries padded to 19*32
#define HDP_  96    // head dim padded to 3*32
#define M_    (B_ * S_)   // 9232 rows, multiple of 16
#define K_    D_
#define N_    D_
#define SCALE_ 0.1118033988749895f  // 1/sqrt(80)

// ---------------------------------------------------------------------------
// Fragment loaders (ISA 7.12.2 layouts), bf16 sources.
// A (16x32): lane half 0 -> K {0..7,16..23}; half 1 -> K {8..15,24..31}.
//            caller passes p = row + k0 + 8*hf.
// B (32x16): same K split as A (half 0 -> {0..7,16..23}, half 1 -> {8..15,24..31}).
//            caller passes p = Wrow + k0 + 8*hf.
// ---------------------------------------------------------------------------
__device__ __forceinline__ v16bf load_afrag(const __bf16* p) {
  v16bf r;
#pragma unroll
  for (int e = 0; e < 8; ++e) r[e] = p[e];
#pragma unroll
  for (int e = 0; e < 8; ++e) r[8 + e] = p[16 + e];
  return r;
}
__device__ __forceinline__ v16bf load_bfrag(const __bf16* p) {
  v16bf r;
#pragma unroll
  for (int e = 0; e < 16; ++e) r[e] = p[e < 8 ? e : e + 8];
  return r;
}

// ---------------------------------------------------------------------------
// Elementwise fp32 -> bf16 pack (8 elements per thread).
// ---------------------------------------------------------------------------
__global__ __launch_bounds__(256)
void f32_to_bf16(const float* __restrict__ src, __bf16* __restrict__ dst,
                 long n) {
  const long i = ((long)blockIdx.x * blockDim.x + threadIdx.x) * 8;
  if (i + 8 <= n) {
    v8bf r;
#pragma unroll
    for (int e = 0; e < 8; ++e) r[e] = (__bf16)src[i + e];
    *(v8bf*)(dst + i) = r;
  }
}

// ---------------------------------------------------------------------------
// WMMA GEMM: out[M_,N_] = A[M_,K_](bf16) @ W[N_,K_](bf16)^T + bias.
// Per-wave tile 32(M) x 64(N): 2 A-frags, 4 B-frags, 8 WMMAs per k-step.
// Block = 256 threads = 8 waves -> 128x128 block tile.  Double-buffered
// k-loop with peeled tail so loads overlap the WMMA chain.
// ---------------------------------------------------------------------------
__global__ __launch_bounds__(256)
void wmma_gemm(const __bf16* __restrict__ A, const __bf16* __restrict__ W,
               const float* __restrict__ bias, float* __restrict__ out) {
  const int wave = threadIdx.x >> 5;
  const int lane = threadIdx.x & 31;
  const int hf   = lane >> 4;
  const int l16  = lane & 15;

  const int  wm = wave & 3, wn = wave >> 2;
  const long m0 = (long)blockIdx.x * 128 + (long)wm * 32;
  const long n0 = (long)blockIdx.y * 128 + (long)wn * 64;
  if (m0 >= M_) return;
  const bool va1 = (m0 + 32) <= M_;   // second 16-row fragment valid?

  const __bf16* arow0 = A + (m0 + l16) * (long)K_ + 8 * hf;
  const __bf16* arow1 = va1 ? arow0 + 16 * (long)K_ : arow0;  // safe dup
  const __bf16* brow0 = W + (n0 +  0 + l16) * (long)K_ + 8 * hf;
  const __bf16* brow1 = W + (n0 + 16 + l16) * (long)K_ + 8 * hf;
  const __bf16* brow2 = W + (n0 + 32 + l16) * (long)K_ + 8 * hf;
  const __bf16* brow3 = W + (n0 + 48 + l16) * (long)K_ + 8 * hf;

  v8f acc0[4] = {}, acc1[4] = {};

  // prime the pipeline
  v16bf aN0 = load_afrag(arow0);
  v16bf aN1 = load_afrag(arow1);
  v16bf bN0 = load_bfrag(brow0);
  v16bf bN1 = load_bfrag(brow1);
  v16bf bN2 = load_bfrag(brow2);
  v16bf bN3 = load_bfrag(brow3);

#pragma unroll 2
  for (int k0 = 0; k0 < K_ - 32; k0 += 32) {
    const v16bf a0 = aN0, a1 = aN1;
    const v16bf b0 = bN0, b1 = bN1, b2 = bN2, b3 = bN3;
    const int kn = k0 + 32;
    aN0 = load_afrag(arow0 + kn);
    aN1 = load_afrag(arow1 + kn);
    bN0 = load_bfrag(brow0 + kn);
    bN1 = load_bfrag(brow1 + kn);
    bN2 = load_bfrag(brow2 + kn);
    bN3 = load_bfrag(brow3 + kn);
    acc0[0] = __builtin_amdgcn_wmma_f32_16x16x32_bf16(false, a0, false, b0, (short)0, acc0[0], false, false);
    acc1[0] = __builtin_amdgcn_wmma_f32_16x16x32_bf16(false, a1, false, b0, (short)0, acc1[0], false, false);
    acc0[1] = __builtin_amdgcn_wmma_f32_16x16x32_bf16(false, a0, false, b1, (short)0, acc0[1], false, false);
    acc1[1] = __builtin_amdgcn_wmma_f32_16x16x32_bf16(false, a1, false, b1, (short)0, acc1[1], false, false);
    acc0[2] = __builtin_amdgcn_wmma_f32_16x16x32_bf16(false, a0, false, b2, (short)0, acc0[2], false, false);
    acc1[2] = __builtin_amdgcn_wmma_f32_16x16x32_bf16(false, a1, false, b2, (short)0, acc1[2], false, false);
    acc0[3] = __builtin_amdgcn_wmma_f32_16x16x32_bf16(false, a0, false, b3, (short)0, acc0[3], false, false);
    acc1[3] = __builtin_amdgcn_wmma_f32_16x16x32_bf16(false, a1, false, b3, (short)0, acc1[3], false, false);
  }
  // peeled last k-step
  acc0[0] = __builtin_amdgcn_wmma_f32_16x16x32_bf16(false, aN0, false, bN0, (short)0, acc0[0], false, false);
  acc1[0] = __builtin_amdgcn_wmma_f32_16x16x32_bf16(false, aN1, false, bN0, (short)0, acc1[0], false, false);
  acc0[1] = __builtin_amdgcn_wmma_f32_16x16x32_bf16(false, aN0, false, bN1, (short)0, acc0[1], false, false);
  acc1[1] = __builtin_amdgcn_wmma_f32_16x16x32_bf16(false, aN1, false, bN1, (short)0, acc1[1], false, false);
  acc0[2] = __builtin_amdgcn_wmma_f32_16x16x32_bf16(false, aN0, false, bN2, (short)0, acc0[2], false, false);
  acc1[2] = __builtin_amdgcn_wmma_f32_16x16x32_bf16(false, aN1, false, bN2, (short)0, acc1[2], false, false);
  acc0[3] = __builtin_amdgcn_wmma_f32_16x16x32_bf16(false, aN0, false, bN3, (short)0, acc0[3], false, false);
  acc1[3] = __builtin_amdgcn_wmma_f32_16x16x32_bf16(false, aN1, false, bN3, (short)0, acc1[3], false, false);

  // epilogue: C layout = lane column, element j -> row j + 8*hf
#pragma unroll
  for (int t = 0; t < 4; ++t) {
    const long n  = n0 + t * 16 + l16;
    const float bv = bias[n];
#pragma unroll
    for (int j = 0; j < 8; ++j)
      out[(m0 + j + 8 * hf) * (long)N_ + n] = acc0[t][j] + bv;
    if (va1) {
#pragma unroll
      for (int j = 0; j < 8; ++j)
        out[(m0 + 16 + j + 8 * hf) * (long)N_ + n] = acc1[t][j] + bv;
    }
  }
}

// ---------------------------------------------------------------------------
// Prep: rotary on q,k (q pre-scaled by 1/sqrt(HD)), convert to bf16, repack:
// qb/kb -> [B,H,SP_,HDP_] (zero padded), vT -> [B,H,HD_,SP_] (transposed).
// ---------------------------------------------------------------------------
__global__ __launch_bounds__(128)
void prep_qkv(const float* __restrict__ qt, const float* __restrict__ kt,
              const float* __restrict__ vt, const float* __restrict__ fcos,
              const float* __restrict__ fsin, __bf16* __restrict__ qb,
              __bf16* __restrict__ kb, __bf16* __restrict__ vT) {
  int idx = blockIdx.x;
  const int s = idx % SP_; idx /= SP_;
  const int h = idx % H_;
  const int b = idx / H_;
  const int d = threadIdx.x;

  const long row = (long)(b * H_ + h);
  if (d < HDP_) {
    __bf16 qv = (__bf16)0.0f, kv = (__bf16)0.0f;
    if (s < S_ && d < HD_) {
      const long src = ((long)(b * S_ + s)) * D_ + h * HD_ + d;
      const float q  = qt[src], k = kt[src];
      const int pair = d >> 1;
      const float c  = fcos[s * HALF_ + pair];
      const float sn = fsin[s * HALF_ + pair];
      const long srcp = src + ((d & 1) ? -1 : 1);
      const float qp = qt[srcp], kp = kt[srcp];
      float qo, ko;
      if ((d & 1) == 0) { qo = q * c - qp * sn; ko = k * c - kp * sn; }
      else              { qo = qp * sn + q * c; ko = kp * sn + k * c; }
      qv = (__bf16)(qo * SCALE_);
      kv = (__bf16)ko;
    }
    const long dst = (row * SP_ + s) * HDP_ + d;
    qb[dst] = qv;
    kb[dst] = kv;
  }
  if (d < HD_) {
    __bf16 vv = (__bf16)0.0f;
    if (s < S_) vv = (__bf16)vt[((long)(b * S_ + s)) * D_ + h * HD_ + d];
    vT[(row * HD_ + d) * SP_ + s] = vv;
  }
}

// ---------------------------------------------------------------------------
// Flash attention: one wave per 16-query tile of one (b,h), streaming 32-key
// chunks.  6 WMMAs for scores (K=96), fp32 online softmax, probs bounced
// C-layout -> LDS -> A-layout, 5 WMMAs for P@V.  V-fragments loaded before
// the softmax so they overlap the VALU work.  Output stored as bf16.
// ---------------------------------------------------------------------------
#define QT_TILES 37                 // ceil(577/16)
__global__ __launch_bounds__(128)
void attn_fa(const __bf16* __restrict__ qb, const __bf16* __restrict__ kb,
             const __bf16* __restrict__ vT, __bf16* __restrict__ attn) {
  __shared__ __bf16 plds[4][16 * 32];

  const int wave = threadIdx.x >> 5;
  const int lane = threadIdx.x & 31;
  const int hf   = lane >> 4;
  const int l16  = lane & 15;

  const int bh = blockIdx.y;                 // 0..255
  const int h  = bh % H_;
  const int b  = bh / H_;
  const int qt = blockIdx.x * 4 + wave;      // 0..39 (37 real tiles)
  const int qtile = qt >= QT_TILES ? QT_TILES - 1 : qt;  // uniform barriers
  const long q0 = (long)qtile * 16;

  const float NEG_INF = -__builtin_inff();

  v16bf qf[3];
  {
    const __bf16* qbase = qb + ((long)bh * SP_ + q0 + l16) * HDP_ + 8 * hf;
#pragma unroll
    for (int ks = 0; ks < 3; ++ks) qf[ks] = load_afrag(qbase + ks * 32);
  }

  v8f o[5] = {};
  float mrow[8], lrow[8];
#pragma unroll
  for (int j = 0; j < 8; ++j) { mrow[j] = NEG_INF; lrow[j] = 0.0f; }

  const __bf16* kbh = kb + (long)bh * SP_ * HDP_;
  const __bf16* vbh = vT + (long)bh * HD_ * SP_;

  for (int kc = 0; kc < SP_; kc += 32) {
    // ---- scores: two 16x16 C tiles over the 32-key chunk
    v8f sc[2] = {};
#pragma unroll
    for (int t = 0; t < 2; ++t) {
      const __bf16* kp = kbh + (long)(kc + t * 16 + l16) * HDP_ + 8 * hf;
#pragma unroll
      for (int ks = 0; ks < 3; ++ks) {
        const v16bf bf = load_bfrag(kp + ks * 32);
        sc[t] = __builtin_amdgcn_wmma_f32_16x16x32_bf16(
            false, qf[ks], false, bf, (short)0, sc[t], false, false);
      }
    }
    // ---- V fragments early: overlap their latency with the softmax VALU
    v16bf vf[5];
#pragma unroll
    for (int t = 0; t < 5; ++t)
      vf[t] = load_bfrag(vbh + (long)(t * 16 + l16) * SP_ + kc + 8 * hf);

    // ---- mask padded keys (scale already folded into q)
    const bool v0 = (kc + l16) < S_;
    const bool v1 = (kc + 16 + l16) < S_;
#pragma unroll
    for (int j = 0; j < 8; ++j) {
      sc[0][j] = v0 ? sc[0][j] : NEG_INF;
      sc[1][j] = v1 ? sc[1][j] : NEG_INF;
    }
    // ---- online softmax (row reductions across 16-lane half groups)
    float cmax[8];
#pragma unroll
    for (int j = 0; j < 8; ++j) cmax[j] = fmaxf(sc[0][j], sc[1][j]);
#pragma unroll
    for (int off = 1; off < 16; off <<= 1)
#pragma unroll
      for (int j = 0; j < 8; ++j)
        cmax[j] = fmaxf(cmax[j], __shfl_xor(cmax[j], off, 32));

    float alpha[8], psum[8];
#pragma unroll
    for (int j = 0; j < 8; ++j) {
      const float mnew = fmaxf(mrow[j], cmax[j]);
      alpha[j] = __expf(mrow[j] - mnew);
      const float p0 = __expf(sc[0][j] - mnew);
      const float p1 = __expf(sc[1][j] - mnew);
      sc[0][j] = p0; sc[1][j] = p1;
      psum[j] = p0 + p1;
      mrow[j] = mnew;
    }
#pragma unroll
    for (int off = 1; off < 16; off <<= 1)
#pragma unroll
      for (int j = 0; j < 8; ++j) psum[j] += __shfl_xor(psum[j], off, 32);
#pragma unroll
    for (int j = 0; j < 8; ++j) lrow[j] = lrow[j] * alpha[j] + psum[j];
#pragma unroll
    for (int t = 0; t < 5; ++t)
#pragma unroll
      for (int j = 0; j < 8; ++j) o[t][j] *= alpha[j];

    // ---- C-layout probs -> row-major LDS -> A-fragment
    __syncthreads();
    __bf16* pl = plds[wave];
#pragma unroll
    for (int t = 0; t < 2; ++t)
#pragma unroll
      for (int j = 0; j < 8; ++j)
        pl[(j + 8 * hf) * 32 + t * 16 + l16] = (__bf16)sc[t][j];
    __syncthreads();

    const v16bf pa = load_afrag(pl + l16 * 32 + 8 * hf);
#pragma unroll
    for (int t = 0; t < 5; ++t)
      o[t] = __builtin_amdgcn_wmma_f32_16x16x32_bf16(
          false, pa, false, vf[t], (short)0, o[t], false, false);
  }

  // ---- finalize + store bf16 attn in [B,S,D] for the output projection
  if (qt < QT_TILES) {
#pragma unroll
    for (int t = 0; t < 5; ++t) {
#pragma unroll
      for (int j = 0; j < 8; ++j) {
        const long row = q0 + j + 8 * hf;
        if (row < S_) {
          attn[((long)b * S_ + row) * D_ + h * HD_ + t * 16 + l16] =
              (__bf16)(o[t][j] / lrow[j]);
        }
      }
    }
  }
}

// ---------------------------------------------------------------------------
extern "C" void kernel_launch(void* const* d_in, const int* in_sizes, int n_in,
                              void* d_out, int out_size, void* d_ws,
                              size_t ws_size, hipStream_t stream) {
  const float* hidden = (const float*)d_in[0];
  const float* fcos   = (const float*)d_in[1];
  const float* fsin   = (const float*)d_in[2];
  const float* Wq     = (const float*)d_in[3];
  const float* bq     = (const float*)d_in[4];
  const float* Wk     = (const float*)d_in[5];
  const float* bk     = (const float*)d_in[6];
  const float* Wv     = (const float*)d_in[7];
  const float* bv     = (const float*)d_in[8];
  const float* Wo     = (const float*)d_in[9];
  const float* bo     = (const float*)d_in[10];
  float* out = (float*)d_out;

  char* ws = (char*)d_ws;
  const size_t WB  = (size_t)D_ * D_ * sizeof(__bf16);           // 3.3 MB
  const size_t HB  = (size_t)M_ * D_ * sizeof(__bf16);           // 23.6 MB
  const size_t TMP = (size_t)M_ * D_ * sizeof(float);            // 47.3 MB
  const size_t QKB = (size_t)B_ * H_ * SP_ * HDP_ * sizeof(__bf16);

  __bf16* Wq_b  = (__bf16*)(ws);
  __bf16* Wk_b  = (__bf16*)(ws + WB);
  __bf16* Wv_b  = (__bf16*)(ws + 2 * WB);
  __bf16* Wo_b  = (__bf16*)(ws + 3 * WB);
  __bf16* hid_b = (__bf16*)(ws + 4 * WB);        // reused for bf16 attn
  float*  q_tmp = (float*)(ws + 4 * WB + HB);
  float*  k_tmp = (float*)(ws + 4 * WB + HB + TMP);
  float*  v_tmp = (float*)(ws + 4 * WB + HB + 2 * TMP);
  __bf16* qbuf  = (__bf16*)(ws + 4 * WB + HB + 3 * TMP);
  __bf16* kbuf  = (__bf16*)(ws + 4 * WB + HB + 3 * TMP + QKB);
  __bf16* vTbuf = (__bf16*)(ws + 4 * WB + HB + 3 * TMP + 2 * QKB);

  // one-time bf16 packs (compute-bound workload; this traffic is noise)
  const long nH = (long)M_ * D_, nW = (long)D_ * D_;
  f32_to_bf16<<<dim3(nH / 2048), dim3(256), 0, stream>>>(hidden, hid_b, nH);
  f32_to_bf16<<<dim3(nW / 2048), dim3(256), 0, stream>>>(Wq, Wq_b, nW);
  f32_to_bf16<<<dim3(nW / 2048), dim3(256), 0, stream>>>(Wk, Wk_b, nW);
  f32_to_bf16<<<dim3(nW / 2048), dim3(256), 0, stream>>>(Wv, Wv_b, nW);
  f32_to_bf16<<<dim3(nW / 2048), dim3(256), 0, stream>>>(Wo, Wo_b, nW);

  const dim3 gblk(256);
  const dim3 ggrid((M_ + 127) / 128, N_ / 128);   // 73 x 10

  wmma_gemm<<<ggrid, gblk, 0, stream>>>(hid_b, Wq_b, bq, q_tmp);
  wmma_gemm<<<ggrid, gblk, 0, stream>>>(hid_b, Wk_b, bk, k_tmp);
  wmma_gemm<<<ggrid, gblk, 0, stream>>>(hid_b, Wv_b, bv, v_tmp);

  prep_qkv<<<dim3(B_ * H_ * SP_), dim3(128), 0, stream>>>(
      q_tmp, k_tmp, v_tmp, fcos, fsin, qbuf, kbuf, vTbuf);

  // hid_b is dead after the V projection: reuse it for bf16 attention output
  attn_fa<<<dim3((QT_TILES + 3) / 4, B_ * H_), dim3(128), 0, stream>>>(
      qbuf, kbuf, vTbuf, hid_b);

  wmma_gemm<<<ggrid, gblk, 0, stream>>>(hid_b, Wo_b, bo, out);
}
